// Transform_78623671320837
// MI455X (gfx1250) — matrix operand order repaired and checked by reading, not compiled, not run on hardware
//
#include <hip/hip_runtime.h>
#include <hip/hip_fp16.h>

typedef __attribute__((ext_vector_type(16))) _Float16 v16h;
typedef __attribute__((ext_vector_type(8)))  float    v8f;

#define N_ITEMS 50000
#define KC 64                 // classes
#define WN 64                 // workers
#define CC (KC*WN)            // 4096 fused output columns (j*64 + w)
#define MT 16                 // rows of f_score per block
#define NWAVES 16
#define THREADS (NWAVES*32)

// ---------------------------------------------------------------------------
// Kernel 1: row-softmax of worker_abi[w,k,:] and pack result into WMMA
// B-fragment order (f16).  B[k, c] with c = j*WN + w.
// Fragment (ct, ks) covers cols [ct*16, ct*16+16), k in [ks*32, ks*32+32).
// Lane layout (16-bit B, contiguous k order used here): lanes 0-15 hold K=ks*32+0..15,
// lanes 16-31 hold K=ks*32+16..31; element h of the per-lane v16h is K offset h.
// Packed storage: Bp[((ct*2+ks)*32 + lane)*16 + h]
// ---------------------------------------------------------------------------
__global__ void softmax_pack_kernel(const float* __restrict__ wabi,
                                    _Float16* __restrict__ Bp) {
  int t = blockIdx.x * blockDim.x + threadIdx.x;   // one thread per (w,k) row
  if (t >= WN * KC) return;
  int w = t >> 6;
  int k = t & 63;
  const float* row = wabi + ((size_t)(w * KC + k)) * KC;

  float m = -3.4e38f;
  for (int j = 0; j < KC; ++j) m = fmaxf(m, row[j]);
  float s = 0.f;
  for (int j = 0; j < KC; ++j) s += __expf(row[j] - m);
  float inv = 1.f / s;

  int ks      = k >> 5;
  int kin     = k & 31;
  int lane_hi = (kin >> 4) << 4;   // 0 or 16
  int h       = kin & 15;
  for (int j = 0; j < KC; ++j) {
    float p  = __expf(row[j] - m) * inv;
    int   c  = j * WN + w;
    int   ct = c >> 4;
    int   nn = c & 15;
    int   lane = lane_hi + nn;
    size_t idx = ((size_t)(ct * 2 + ks) * 32 + lane) * 16 + h;
    Bp[idx] = (_Float16)p;
  }
}

// ---------------------------------------------------------------------------
// Kernel 2: fused GEMM (16 x 4096 per block, K=64, f16 WMMA, f32 accum)
//           + softmax over j (stride-64 groups of the 4096 columns).
// Wave wv owns cols [wv*256, wv*256+256) = subtiles st=0..15.
// For subtile st: j = wv*4 + (st>>2) is CONSTANT, w = (st&3)*16 + (lane&15).
// Cross-wave softmax reduction through LDS partials.
// Output stores are non-temporal: 1.64 GB of write-once streaming data must
// not churn L2, where the 512 KB packed-B table and A rows should stay hot.
// ---------------------------------------------------------------------------
__launch_bounds__(THREADS, 1)
__global__ void fused_gemm_softmax_kernel(const float* __restrict__ A,
                                          const _Float16* __restrict__ Bp,
                                          float* __restrict__ out) {
  __shared__ float red[NWAVES * MT * WN];  // 16*16*64 f32 = 64 KB partials
  __shared__ float fbuf[MT * WN];          // final per-(n,w) max
  __shared__ float sbuf[MT * WN];          // final per-(n,w) sum

  const int tid     = threadIdx.x;
  const int lane    = tid & 31;
  const int wv      = tid >> 5;
  const int n       = lane & 15;
  const int hf      = lane >> 4;
  const int rowbase = blockIdx.x * MT;

  // ---- A fragments (fp32 -> f16), contiguous-k 16-bit A 16x32 layout (matches packed B):
  // lane (hf,n): VGPR pair p holds K = 2p + 16*hf (+ks*32)
  v16h a0, a1;
  {
    const float* arow = A + (size_t)(rowbase + n) * KC;
#pragma unroll
    for (int p = 0; p < 8; ++p) {
      int kb = 2 * p + 16 * hf;
      float2 f0 = *(const float2*)(arow + kb);
      float2 f1 = *(const float2*)(arow + kb + 32);
      a0[2 * p]     = (_Float16)f0.x;
      a0[2 * p + 1] = (_Float16)f0.y;
      a1[2 * p]     = (_Float16)f1.x;
      a1[2 * p + 1] = (_Float16)f1.y;
    }
  }

  // ---- GEMM: 16 subtiles x 2 k-steps of v_wmma_f32_16x16x32_f16
  const v16h* bfr = (const v16h*)Bp;
  v8f acc[16];
#pragma unroll
  for (int st = 0; st < 16; ++st) {
    int frag = (wv * 16 + st) * 2;
    v16h b0 = bfr[(size_t)frag * 32 + lane];
    v16h b1 = bfr[(size_t)(frag + 1) * 32 + lane];
    v8f c = {};
    c = __builtin_amdgcn_wmma_f32_16x16x32_f16(false, a0, false, b0,
                                               (short)0, c, false, false);
    c = __builtin_amdgcn_wmma_f32_16x16x32_f16(false, a1, false, b1,
                                               (short)0, c, false, false);
    acc[st] = c;
  }

  // ---- Phase 1: per-wave partial max over this wave's 4 j values
  // acc[st][r]: row M = hf*8 + r, col = wv*256 + st*16 + n
#pragma unroll
  for (int r = 0; r < 8; ++r) {
    int M = hf * 8 + r;
#pragma unroll
    for (int sm = 0; sm < 4; ++sm) {
      float m = acc[sm][r];
      m = fmaxf(m, acc[sm + 4][r]);
      m = fmaxf(m, acc[sm + 8][r]);
      m = fmaxf(m, acc[sm + 12][r]);
      red[wv * (MT * WN) + (M << 6) + sm * 16 + n] = m;
    }
  }
  __syncthreads();

  // ---- reduce partial maxes across the 16 waves -> fbuf
  for (int idx = tid; idx < MT * WN; idx += THREADS) {
    float m = red[idx];
#pragma unroll
    for (int i = 1; i < NWAVES; ++i) m = fmaxf(m, red[i * (MT * WN) + idx]);
    fbuf[idx] = m;
  }
  __syncthreads();

  // ---- write raw ob_labels (first half of d_out), non-temporal streaming
#pragma unroll
  for (int st = 0; st < 16; ++st) {
    size_t col = (size_t)wv * 256 + st * 16 + n;
#pragma unroll
    for (int r = 0; r < 8; ++r) {
      int M = hf * 8 + r;
      __builtin_nontemporal_store(acc[st][r],
                                  &out[(size_t)(rowbase + M) * CC + col]);
    }
  }

  // ---- Phase 2: exp(x - max) in place, per-wave partial sums
#pragma unroll
  for (int sm = 0; sm < 4; ++sm) {
#pragma unroll
    for (int r = 0; r < 8; ++r) {
      int M = hf * 8 + r;
      float fm = fbuf[(M << 6) + sm * 16 + n];
      float ps = 0.f;
#pragma unroll
      for (int jl = 0; jl < 4; ++jl) {
        int   st = jl * 4 + sm;
        float e  = __expf(acc[st][r] - fm);
        acc[st][r] = e;
        ps += e;
      }
      red[wv * (MT * WN) + (M << 6) + sm * 16 + n] = ps;
    }
  }
  __syncthreads();

  // ---- reduce partial sums across the 16 waves -> sbuf
  for (int idx = tid; idx < MT * WN; idx += THREADS) {
    float s = red[idx];
#pragma unroll
    for (int i = 1; i < NWAVES; ++i) s += red[i * (MT * WN) + idx];
    sbuf[idx] = s;
  }
  __syncthreads();

  // ---- write ob_labels_soft (second half of d_out), non-temporal streaming
  float* soft = out + (size_t)N_ITEMS * CC;
#pragma unroll
  for (int sm = 0; sm < 4; ++sm) {
#pragma unroll
    for (int r = 0; r < 8; ++r) {
      int   M   = hf * 8 + r;
      float inv = 1.0f / sbuf[(M << 6) + sm * 16 + n];
#pragma unroll
      for (int jl = 0; jl < 4; ++jl) {
        int    st  = jl * 4 + sm;
        size_t col = (size_t)wv * 256 + st * 16 + n;
        __builtin_nontemporal_store(acc[st][r] * inv,
                                    &soft[(size_t)(rowbase + M) * CC + col]);
      }
    }
  }
}

extern "C" void kernel_launch(void* const* d_in, const int* in_sizes, int n_in,
                              void* d_out, int out_size, void* d_ws, size_t ws_size,
                              hipStream_t stream) {
  const float* f_score    = (const float*)d_in[0];   // [N, K] fp32
  const float* worker_abi = (const float*)d_in[1];   // [W, K, K] fp32
  float*       out        = (float*)d_out;           // [N,K,W] + [N,K,W] fp32
  _Float16*    Bp         = (_Float16*)d_ws;         // 512 KB packed B fragments

  // softmax + pack B: 4096 rows, one thread each
  softmax_pack_kernel<<<(WN * KC) / 64, 64, 0, stream>>>(worker_abi, Bp);

  // fused GEMM + softmax: 50000/16 = 3125 blocks of 512 threads
  fused_gemm_softmax_kernel<<<N_ITEMS / MT, THREADS, 0, stream>>>(f_score, Bp, out);
}
